// Net_54898271978111
// MI455X (gfx1250) — compile-verified, matrix operand order checked by reading, not run on hardware
//
#include <hip/hip_runtime.h>

typedef _Float16 v16h __attribute__((ext_vector_type(16)));
typedef float    v8f  __attribute__((ext_vector_type(8)));

#define ACT_Q 2.0f

// partial_activation(Y, n=16, m=2, t=2) on one row of 32 values (in registers).
// Masks Y in place; returns per-row train contribution and success flag.
__device__ __forceinline__ void partial_act(float* Y, float& t_row, int& flag) {
  int   minI = 0, maxI = 0;
  float minv = 3.0f, maxv = -1.0f;
  float ymin0 = 0.f, ymin1 = 0.f, ymax0 = 0.f, ymax1 = 0.f;
#pragma unroll
  for (int bi = 0; bi < 16; ++bi) {
    float y0 = Y[2 * bi], y1 = Y[2 * bi + 1];
    float pos = (y0 > 0.0f ? 1.0f : 0.0f) + (y1 > 0.0f ? 1.0f : 0.0f);
    if (pos < minv) { minv = pos; minI = bi; ymin0 = y0; ymin1 = y1; }
    if (pos > maxv) { maxv = pos; maxI = bi; ymax0 = y0; ymax1 = y1; }
  }
  const bool max_succ = (maxv == 2.0f);
  const bool min_succ = (minv == 0.0f);
  // train terms from ORIGINAL Y values
  float cmin = (ymin0 > 0.f ? ymin0 : 0.f) + (ymin1 > 0.f ? ymin1 : 0.f);
  float cmax = -((ymax0 < 0.f ? ymax0 : 0.f) + (ymax1 < 0.f ? ymax1 : 0.f));
  t_row = (min_succ ? 0.0f : cmin) + (max_succ ? 0.0f : cmax);
  flag  = (min_succ && max_succ) ? 1 : 0;
  // mask: scale max block by Q, then zero min block (min overrides max)
#pragma unroll
  for (int bi = 0; bi < 16; ++bi) {
    float s = (bi == maxI && max_succ) ? ACT_Q : 1.0f;
    if (bi == minI && min_succ) s = 0.0f;
    Y[2 * bi] *= s;
    Y[2 * bi + 1] *= s;
  }
}

__global__ __launch_bounds__(256)
void net_fused_kernel(const float* __restrict__ x,
                      const float* __restrict__ W1, const float* __restrict__ b1,
                      const float* __restrict__ W2, const float* __restrict__ b2,
                      const float* __restrict__ W3, const float* __restrict__ b3,
                      float* __restrict__ out, float* __restrict__ partials) {
  __shared__ _Float16 h1s[256 * 32];  // 16 KB : layer-1 activations (f16, WMMA A source)
  __shared__ float    y2s[256 * 32];  // 32 KB : layer-2 results (f32 D fragments)
  __shared__ float    redS[256];
  __shared__ float    redT[256];

  const int tid  = threadIdx.x;
  const int lane = tid & 31;
  const int wave = tid >> 5;          // 8 waves, each owns 32 consecutive rows
  const int lrow = lane & 15;
  const int hi   = lane >> 4;         // 0: k 0-7,16-23; 1: k 8-15,24-31 (ISA A/B layout)
  const long g   = (long)blockIdx.x * 256 + tid;

  // ---------------- Layer 1: Y1 = x @ W1^T + b1  (K=2, scalar f32) -------------
  const float x0 = x[2 * g], x1 = x[2 * g + 1];
  float Y[32];
#pragma unroll
  for (int j = 0; j < 32; ++j)
    Y[j] = fmaf(x0, W1[2 * j], fmaf(x1, W1[2 * j + 1], b1[j]));

  float t1; int f1;
  partial_act(Y, t1, f1);

  // stash h1 as f16 for the WMMA A-fragment gather
#pragma unroll
  for (int j = 0; j < 32; ++j) h1s[tid * 32 + j] = (_Float16)Y[j];
  __syncthreads();

  // ---------------- Layer 2: Y2 = h1 @ W2^T + b2  via v_wmma_f32_16x16x32_f16 --
  // Per wave: 2 A-tiles (rows 0-15 / 16-31 of its chunk) x 2 N-tiles = 4 WMMAs.
#pragma unroll
  for (int tn = 0; tn < 2; ++tn) {
    // B fragment: B[k][n] = W2[n][k]; lane holds column n = tn*16 + lrow.
    v16h bf;
    const int n = tn * 16 + lrow;
#pragma unroll
    for (int v = 0; v < 8; ++v) {
      const int kb = (v < 4 ? 2 * v : 16 + 2 * (v - 4)) + 8 * hi;
      bf[2 * v]     = (_Float16)W2[n * 32 + kb];
      bf[2 * v + 1] = (_Float16)W2[n * 32 + kb + 1];
    }
    const float cval = b2[n];  // bias broadcast down the M dimension

#pragma unroll
    for (int tb = 0; tb < 2; ++tb) {
      // A fragment per ISA 16-bit 16x32 layout
      v16h af;
      const int rowg = wave * 32 + tb * 16 + lrow;
#pragma unroll
      for (int v = 0; v < 8; ++v) {
        const int kb = (v < 4 ? 2 * v : 16 + 2 * (v - 4)) + 8 * hi;
        af[2 * v]     = h1s[rowg * 32 + kb];
        af[2 * v + 1] = h1s[rowg * 32 + kb + 1];
      }
      v8f c;
#pragma unroll
      for (int p = 0; p < 8; ++p) c[p] = cval;

      v8f d = __builtin_amdgcn_wmma_f32_16x16x32_f16(
          /*neg_a=*/false, af, /*neg_b=*/false, bf,
          /*c_mod=*/(short)0, c, /*reuse_a=*/false, /*reuse_b=*/false);

      // D layout: VGPR p -> row M = p + 8*hi, col N = lrow (within tile)
#pragma unroll
      for (int p = 0; p < 8; ++p)
        y2s[(wave * 32 + tb * 16 + p + 8 * hi) * 32 + tn * 16 + lrow] = d[p];
    }
  }
  __syncthreads();

#pragma unroll
  for (int j = 0; j < 32; ++j) Y[j] = y2s[tid * 32 + j];

  float t2; int f2;
  partial_act(Y, t2, f2);

  // ---------------- Layer 3 + softmax (2 outputs) ------------------------------
  float o0 = b3[0], o1 = b3[1];
#pragma unroll
  for (int k = 0; k < 32; ++k) {
    o0 = fmaf(Y[k], W3[k], o0);
    o1 = fmaf(Y[k], W3[32 + k], o1);
  }
  const float m  = fmaxf(o0, o1);
  const float e0 = __expf(o0 - m), e1 = __expf(o1 - m);
  const float inv = 1.0f / (e0 + e1);
  out[2 * g]     = e0 * inv;
  out[2 * g + 1] = e1 * inv;

  // ---------------- Deterministic block reduction (succ count, train sum) ------
  redS[tid] = (float)(f1 + f2);
  redT[tid] = t1 + t2;
  __syncthreads();
#pragma unroll
  for (int s = 128; s > 0; s >>= 1) {
    if (tid < s) { redS[tid] += redS[tid + s]; redT[tid] += redT[tid + s]; }
    __syncthreads();
  }
  if (tid == 0) {
    partials[2 * blockIdx.x]     = redS[0];
    partials[2 * blockIdx.x + 1] = redT[0];
  }
}

__global__ __launch_bounds__(256)
void reduce_kernel(const float* __restrict__ partials, int nblocks,
                   float* __restrict__ scal, float succScale) {
  __shared__ float rs[256], rt[256];
  const int tid = threadIdx.x;
  float s = 0.f, t = 0.f;
  for (int i = tid; i < nblocks; i += 256) {
    s += partials[2 * i];
    t += partials[2 * i + 1];
  }
  rs[tid] = s; rt[tid] = t;
  __syncthreads();
#pragma unroll
  for (int k = 128; k > 0; k >>= 1) {
    if (tid < k) { rs[tid] += rs[tid + k]; rt[tid] += rt[tid + k]; }
    __syncthreads();
  }
  if (tid == 0) {
    scal[0] = rs[0] * succScale;  // (s1+s2)/2 with s = mean over B  => sum/(2B)
    scal[1] = rt[0];              // train = t1 + t2
  }
}

extern "C" void kernel_launch(void* const* d_in, const int* in_sizes, int n_in,
                              void* d_out, int out_size, void* d_ws, size_t ws_size,
                              hipStream_t stream) {
  const float* x  = (const float*)d_in[0];
  const float* W1 = (const float*)d_in[1];
  const float* b1 = (const float*)d_in[2];
  const float* W2 = (const float*)d_in[3];
  const float* b2 = (const float*)d_in[4];
  const float* W3 = (const float*)d_in[5];
  const float* b3 = (const float*)d_in[6];

  float* out      = (float*)d_out;
  float* partials = (float*)d_ws;

  const int B       = in_sizes[0] / 2;  // x is [B, 2]
  const int nblocks = B / 256;          // B = 1<<20 -> 4096 blocks

  net_fused_kernel<<<nblocks, 256, 0, stream>>>(x, W1, b1, W2, b2, W3, b3,
                                                out, partials);
  reduce_kernel<<<1, 256, 0, stream>>>(partials, nblocks,
                                       out + (size_t)2 * B,
                                       1.0f / (2.0f * (float)B));
}
